// SoftmaxSetAttention_91036126806074
// MI455X (gfx1250) — matrix operand order repaired and checked by reading, not compiled, not run on hardware
//
#include <hip/hip_runtime.h>
#include <math.h>

#define Bc 2
#define Hc 16
#define LQc 2048
#define LKc 2048
#define Dc 128
#define KT 32
#define QTILE 128
#define NWAVES 8

typedef __attribute__((ext_vector_type(16))) __bf16 v16bf;
typedef __attribute__((ext_vector_type(8)))  __bf16 v8bf;
typedef __attribute__((ext_vector_type(4)))  __bf16 v4bf;
typedef __attribute__((ext_vector_type(2)))  __bf16 v2bf;
typedef __attribute__((ext_vector_type(8)))  float  v8f;

union V16 { v16bf v; v8bf h[2]; };

// raw v_exp_f32 (base-2 exponential)
__device__ __forceinline__ float exp2_fast(float x) {
  return __builtin_amdgcn_exp2f(x);
}

// xor-lane max within the 16-lane DPP row via ROW_XMASK (VALU, no LDS traffic)
template <int MASK>
__device__ __forceinline__ float rxor_max(float x) {
  int xi = __float_as_int(x);
  int yi = __builtin_amdgcn_update_dpp(xi, xi, 0x160 | MASK, 0xf, 0xf, true);
  return fmaxf(x, __int_as_float(yi));
}
__device__ __forceinline__ float half_max16(float x) {
  x = rxor_max<1>(x);
  x = rxor_max<2>(x);
  x = rxor_max<4>(x);
  x = rxor_max<8>(x);
  return x;
}

__global__ __launch_bounds__(256, 1)
void fa_fwd(const float* __restrict__ Q, const float* __restrict__ K,
            const float* __restrict__ V, const int* __restrict__ KM,
            float* __restrict__ O) {
  // K tile [key][d] bf16, padded stride 136 (conflict-free B-frag reads)
  __shared__ __bf16 Klds[KT][136];
  // V tile transposed [d][staged_key] bf16, padded stride 40
  __shared__ __bf16 Vlds[Dc][40];
  // per-wave P staging tile [row][staged_key], padded stride 40
  __shared__ __bf16 Plds[NWAVES][16][40];
  __shared__ float  Blds[KT];

  const int tid  = threadIdx.x;
  const int lane = tid & 31;
  const int wave = tid >> 5;
  const int lh   = lane & 15;
  const int half = lane >> 4;

  const int b = blockIdx.z, h = blockIdx.y;
  const size_t bh = (size_t)b * Hc + h;
  const float* kb_g = K + bh * (size_t)LKc * Dc;
  const float* vb_g = V + bh * (size_t)LKc * Dc;
  const int qbase = blockIdx.x * QTILE + wave * 16;
  const float* qrow = Q + (bh * (size_t)LQc + qbase + lh) * (size_t)Dc;

  // log2-domain softmax: v_exp_f32/v_log_f32 are natively base-2.
  // x = (q.k)*scale*log2e + log2(mult);  p = exp2(x - m)  ==  base-e softmax.
  const float scale2 = 0.08838834764831845f * 1.4426950408889634f;

  // per-thread cooperative-load coordinates (4 float4 slots each)
  int ckey[4], cd[4];
  #pragma unroll
  for (int i = 0; i < 4; ++i) {
    int f = tid + 256 * i;
    ckey[i] = f >> 5;
    cd[i]   = (f & 31) << 2;
  }

  // ---- load this wave's 16x128 Q tile as 4 bf16 A fragments (16x32 each) ----
  v16bf qa[4];
  #pragma unroll
  for (int c = 0; c < 4; ++c) {
    const float* p0 = qrow + c * 32 + half * 16;
    #pragma unroll
    for (int j = 0; j < 8; ++j) qa[c][j] = (__bf16)p0[j];
    #pragma unroll
    for (int j = 0; j < 8; ++j) qa[c][j + 8] = (__bf16)p0[8 + j];
  }

  // all-ones bf16 B fragment: P x ones accumulates the softmax denominator l
  v16bf onesb;
  #pragma unroll
  for (int j = 0; j < 16; ++j) onesb[j] = (__bf16)1.0f;

  // acc[0..7]: 16x16 output tiles over D=128; acc[8]: l (row-sum) tile
  v8f acc[9];
  #pragma unroll
  for (int t = 0; t < 9; ++t)
    #pragma unroll
    for (int r = 0; r < 8; ++r) acc[t][r] = 0.0f;

  float mrow[8];
  #pragma unroll
  for (int r = 0; r < 8; ++r) mrow[r] = -1e30f;

  // ---- register double-buffer: prefetch tile 0 (f32 -> bf16 at load) ----
  v4bf kr[4], vr[4];
  float breg = 0.0f;
  #pragma unroll
  for (int i = 0; i < 4; ++i) {
    const float4 kq = *(const float4*)(kb_g + (size_t)ckey[i] * Dc + cd[i]);
    kr[i][0] = (__bf16)kq.x; kr[i][1] = (__bf16)kq.y;
    kr[i][2] = (__bf16)kq.z; kr[i][3] = (__bf16)kq.w;
    const float4 vq = *(const float4*)(vb_g + (size_t)ckey[i] * Dc + cd[i]);
    vr[i][0] = (__bf16)vq.x; vr[i][1] = (__bf16)vq.y;
    vr[i][2] = (__bf16)vq.z; vr[i][3] = (__bf16)vq.w;
  }
  if (tid < KT) breg = __log2f((float)KM[(size_t)b * LKc + tid]);

  for (int k0 = 0; k0 < LKc; k0 += KT) {
    // ---- publish staged tile to LDS (V with interleaved key permutation) ----
    #pragma unroll
    for (int i = 0; i < 4; ++i) {
      *(v4bf*)&Klds[ckey[i]][cd[i]] = kr[i];
      int kp = 2 * (ckey[i] & 15) + (ckey[i] >> 4);  // staged key column
      Vlds[cd[i] + 0][kp] = vr[i][0];
      Vlds[cd[i] + 1][kp] = vr[i][1];
      Vlds[cd[i] + 2][kp] = vr[i][2];
      Vlds[cd[i] + 3][kp] = vr[i][3];
    }
    if (tid < KT) Blds[tid] = breg;
    __syncthreads();

    // ---- prefetch next tile into registers (overlaps with compute) ----
    if (k0 + KT < LKc) {
      #pragma unroll
      for (int i = 0; i < 4; ++i) {
        const float4 kq = *(const float4*)(kb_g + (size_t)(k0 + KT + ckey[i]) * Dc + cd[i]);
        kr[i][0] = (__bf16)kq.x; kr[i][1] = (__bf16)kq.y;
        kr[i][2] = (__bf16)kq.z; kr[i][3] = (__bf16)kq.w;
        const float4 vq = *(const float4*)(vb_g + (size_t)(k0 + KT + ckey[i]) * Dc + cd[i]);
        vr[i][0] = (__bf16)vq.x; vr[i][1] = (__bf16)vq.y;
        vr[i][2] = (__bf16)vq.z; vr[i][3] = (__bf16)vq.w;
      }
      if (tid < KT) breg = __log2f((float)KM[(size_t)b * LKc + k0 + KT + tid]);
    }

    // ---- S = Q K^T : two 16x16 tiles, K-dim accumulated over D=128 ----
    v8f s0, s1;
    #pragma unroll
    for (int r = 0; r < 8; ++r) { s0[r] = 0.0f; s1[r] = 0.0f; }
    #pragma unroll
    for (int c = 0; c < 4; ++c) {
      V16 kb0, kb1;
      const __bf16* kp0 = &Klds[lh][c * 32 + half * 16];
      const __bf16* kp1 = &Klds[16 + lh][c * 32 + half * 16];
      kb0.h[0] = *(const v8bf*)(kp0); kb0.h[1] = *(const v8bf*)(kp0 + 8);
      kb1.h[0] = *(const v8bf*)(kp1); kb1.h[1] = *(const v8bf*)(kp1 + 8);
      s0 = __builtin_amdgcn_wmma_f32_16x16x32_bf16(false, qa[c], false, kb0.v,
                                                   (short)0, s0, false, false);
      s1 = __builtin_amdgcn_wmma_f32_16x16x32_bf16(false, qa[c], false, kb1.v,
                                                   (short)0, s1, false, false);
    }

    // ---- online softmax (log2 domain): max via DPP, sum via ones-WMMA ----
    const float b0 = Blds[lh], b1 = Blds[16 + lh];
    float p0[8], p1[8], alpha[8];
    #pragma unroll
    for (int r = 0; r < 8; ++r) {
      float x0 = fmaf(s0[r], scale2, b0);
      float x1 = fmaf(s1[r], scale2, b1);
      float tm = half_max16(fmaxf(x0, x1));
      float mn = fmaxf(mrow[r], tm);
      alpha[r] = exp2_fast(mrow[r] - mn);
      p0[r] = exp2_fast(x0 - mn);
      p1[r] = exp2_fast(x1 - mn);
      mrow[r] = mn;
    }
    #pragma unroll
    for (int t = 0; t < 9; ++t)
      #pragma unroll
      for (int r = 0; r < 8; ++r) acc[t][r] *= alpha[r];

    // ---- stage P (C-layout -> A-layout) packed bf16x2, wave-local LDS ----
    __bf16* pw = &Plds[wave][0][0];
    #pragma unroll
    for (int r = 0; r < 8; ++r) {
      v2bf pk;
      pk[0] = (__bf16)p0[r];  // staged col 2*lh   (key tile 0)
      pk[1] = (__bf16)p1[r];  // staged col 2*lh+1 (key tile 1)
      *(v2bf*)&pw[(8 * half + r) * 40 + 2 * lh] = pk;
    }
    V16 pa;
    const __bf16* pp = &Plds[wave][lh][0];
    pa.h[0] = *(const v8bf*)(pp + half * 16);
    pa.h[1] = *(const v8bf*)(pp + half * 16 + 8);

    // ---- O += P V (8 tiles) and l += P 1 (9th tile) ----
    #pragma unroll
    for (int t = 0; t < 8; ++t) {
      V16 vbf;
      const __bf16* vp = &Vlds[t * 16 + lh][half * 16];
      vbf.h[0] = *(const v8bf*)(vp);
      vbf.h[1] = *(const v8bf*)(vp + 8);
      acc[t] = __builtin_amdgcn_wmma_f32_16x16x32_bf16(false, pa.v, false, vbf.v,
                                                       (short)0, acc[t], false, false);
    }
    acc[8] = __builtin_amdgcn_wmma_f32_16x16x32_bf16(false, pa.v, false, onesb,
                                                     (short)0, acc[8], false, false);
    __syncthreads();
  }

  // ---- normalize by l and write output (row = 8*half + r, col = lh) ----
  float* orow_base = O + (bh * (size_t)LQc + qbase) * (size_t)Dc;
  #pragma unroll
  for (int r = 0; r < 8; ++r) {
    float inv = 1.0f / acc[8][r];
    float* orow = orow_base + (size_t)(8 * half + r) * Dc;
    #pragma unroll
    for (int t = 0; t < 8; ++t) orow[t * 16 + lh] = acc[t][r] * inv;
  }
}

extern "C" void kernel_launch(void* const* d_in, const int* in_sizes, int n_in,
                              void* d_out, int out_size, void* d_ws, size_t ws_size,
                              hipStream_t stream) {
  (void)in_sizes; (void)n_in; (void)out_size; (void)d_ws; (void)ws_size;
  const float* q  = (const float*)d_in[0];
  const float* k  = (const float*)d_in[1];
  const float* v  = (const float*)d_in[2];
  const int*   km = (const int*)d_in[3];
  float* out = (float*)d_out;

  dim3 grid(LQc / QTILE, Hc, Bc);  // (16, 16, 2)
  dim3 block(256);                 // 8 wave32 waves
  fa_fwd<<<grid, block, 0, stream>>>(q, k, v, km, out);
}
